// DGN4_70428873720435
// MI455X (gfx1250) — matrix operand order repaired and checked by reading, not compiled, not run on hardware
//
#include <hip/hip_runtime.h>
#include <hip/hip_bf16.h>
#include <math.h>

typedef __attribute__((ext_vector_type(16))) _Float16 v16h;
typedef __attribute__((ext_vector_type(8)))  _Float16 v8h;
typedef __attribute__((ext_vector_type(8)))  float    v8f;

#define BB 4
#define TT 2048
#define DD 256
#define TOPK 8
#define BOTK 12
#define NPART 16   // partial lists per row (2 lanes x 8 waves)

__device__ __forceinline__ bool top_better(float va, int ia, float vb, int ib) {
  return (va > vb) || (va == vb && ia < ib);   // value desc, index asc (stable top_k)
}
__device__ __forceinline__ bool bot_better(float va, int ia, float vb, int ib) {
  return (va < vb) || (va == vb && ia < ib);   // value asc, index asc
}

// ---------------- Kernel 1: row-normalize x -> f16 xn ----------------
__global__ __launch_bounds__(256) void dgn_normalize(const float* __restrict__ x,
                                                     _Float16* __restrict__ xn) {
  const int row  = blockIdx.x * 8 + (threadIdx.x >> 5);
  const int lane = threadIdx.x & 31;
  const float4* xp = (const float4*)(x + (size_t)row * DD + lane * 8);
  float4 a = xp[0], b = xp[1];
  float ss = a.x*a.x + a.y*a.y + a.z*a.z + a.w*a.w
           + b.x*b.x + b.y*b.y + b.z*b.z + b.w*b.w;
#pragma unroll
  for (int o = 16; o >= 1; o >>= 1) ss += __shfl_xor(ss, o, 32);
  const float inv = 1.0f / fmaxf(sqrtf(ss), 1e-12f);
  v8h h;
  h[0] = (_Float16)(a.x * inv); h[1] = (_Float16)(a.y * inv);
  h[2] = (_Float16)(a.z * inv); h[3] = (_Float16)(a.w * inv);
  h[4] = (_Float16)(b.x * inv); h[5] = (_Float16)(b.y * inv);
  h[6] = (_Float16)(b.z * inv); h[7] = (_Float16)(b.w * inv);
  *(v8h*)(xn + (size_t)row * DD + lane * 8) = h;
}

// ---------------- Kernel 2: WMMA sim tiles + exact top/bot-k + aggregate ----------------
__global__ __launch_bounds__(256, 1) void dgn_main(
    const float* __restrict__ x, const _Float16* __restrict__ xn,
    const float* __restrict__ gain, const float* __restrict__ bias,
    const float* __restrict__ log_mix, const float* __restrict__ log_alpha,
    const float* __restrict__ log_scale, float* __restrict__ out) {
  __shared__ float sTile[8][16][32];
  __shared__ float sTopV[16][NPART * TOPK];
  __shared__ int   sTopI[16][NPART * TOPK];
  __shared__ float sBotV[16][NPART * BOTK];
  __shared__ int   sBotI[16][NPART * BOTK];
  __shared__ int   sSel[16][12];
  __shared__ int   sNS[16];
  __shared__ int   sNC[16];

  const int tid = threadIdx.x;
  const int w   = tid >> 5;
  const int l   = tid & 31;
  const int m   = l & 15;
  const int hi  = l >> 4;

  const int b  = blockIdx.x >> 7;           // T/16 == 128 row-blocks per batch
  const int t0 = (blockIdx.x & 127) << 4;

  // Hoist A fragments: rows t0..t0+15, full K=256 (8 chunks of 32).
  // 16-bit A layout: lanes 0-15 hold K {0..7,16..23}; lanes 16-31 hold K {8..15,24..31}.
  const _Float16* arow = xn + ((size_t)(b * TT + t0 + m)) * DD;
  v16h af[8];
#pragma unroll
  for (int kc = 0; kc < 8; ++kc) {
    union { v16h v; v8h h[2]; } u;
    u.h[0] = *(const v8h*)(arow + kc * 32 + hi * 8);
    u.h[1] = *(const v8h*)(arow + kc * 32 + 16 + hi * 8);
    af[kc] = u.v;
  }

  // Per-lane sorted partial lists (registers, constant indices only).
  float tv[TOPK]; int ti[TOPK];
  float bv[BOTK]; int bi[BOTK];
#pragma unroll
  for (int q = 0; q < TOPK; ++q) { tv[q] = -1e30f; ti[q] = 0x7fffffff; }
#pragma unroll
  for (int q = 0; q < BOTK; ++q) { bv[q] =  1e30f; bi[q] = 0x7fffffff; }

  const int trow = t0 + m;                  // row this lane selects for
  const int ndt  = (t0 >> 5) + 1;           // 32-col double tiles covering s <= t0+15
  const _Float16* xnb = xn + (size_t)b * TT * DD;

  for (int dt = w; dt < ndt; dt += 8) {
    const int s0 = dt << 5;
    // B layout (K x N): lane n holds K {0..7,16..23} of row s0+n, lane n+16 holds K {8..15,24..31}.
    const _Float16* brow0 = xnb + (size_t)(s0 + m) * DD + hi * 8;
    const _Float16* brow1 = brow0 + (size_t)16 * DD;

    // Stage ALL B fragments first (independent dest regs -> one big load clause).
    v16h bf0[8], bf1[8];
#pragma unroll
    for (int kc = 0; kc < 8; ++kc) {
      union { v16h v; v8h h[2]; } u0, u1;
      u0.h[0] = *(const v8h*)(brow0 + kc * 32);
      u0.h[1] = *(const v8h*)(brow0 + kc * 32 + 16);
      u1.h[0] = *(const v8h*)(brow1 + kc * 32);
      u1.h[1] = *(const v8h*)(brow1 + kc * 32 + 16);
      bf0[kc] = u0.v;
      bf1[kc] = u1.v;
    }

    // Pure scheduling fence: forbid the backend from sinking the staged loads
    // down into the WMMA loop (which would recycle registers and force a full
    // s_wait_loadcnt 0 before every WMMA). No hardware wait is emitted; the
    // WMMA loop below gets partial s_wait_loadcnt counts instead.
    __builtin_amdgcn_sched_barrier(0);

    // Interleave the two independent accumulation chains (dependency distance 2)
    // so WMMA->WMMA accumulation latency is hidden.
    v8f acc0 = {}; v8f acc1 = {};
#pragma unroll
    for (int kc = 0; kc < 8; ++kc) {
      acc0 = __builtin_amdgcn_wmma_f32_16x16x32_f16(false, af[kc], false, bf0[kc],
                                                    (short)0, acc0, false, false);
      acc1 = __builtin_amdgcn_wmma_f32_16x16x32_f16(false, af[kc], false, bf1[kc],
                                                    (short)0, acc1, false, false);
    }

    // D layout: VGPR r = row (r + 8*hi), col = lane%16. Scatter to per-wave LDS slab.
#pragma unroll
    for (int r = 0; r < 8; ++r) {
      sTile[w][r + 8 * hi][m]      = acc0[r];
      sTile[w][r + 8 * hi][m + 16] = acc1[r];
    }
    asm volatile("s_wait_dscnt 0" ::: "memory");  // intra-wave LDS transpose fence
    // Selection: 2 lanes per row; this lane scans cols hi*16..hi*16+15 of row m.
#pragma unroll
    for (int j = 0; j < 16; ++j) {
      const int s = s0 + hi * 16 + j;
      if (s >= trow) continue;                    // strictly-past mask
      const float v = sTile[w][m][hi * 16 + j];
      if (top_better(v, s, tv[TOPK - 1], ti[TOPK - 1])) {
        tv[TOPK - 1] = v; ti[TOPK - 1] = s;
#pragma unroll
        for (int q = TOPK - 2; q >= 0; --q)
          if (top_better(tv[q + 1], ti[q + 1], tv[q], ti[q])) {
            float fv = tv[q]; int fi = ti[q];
            tv[q] = tv[q + 1]; ti[q] = ti[q + 1];
            tv[q + 1] = fv;    ti[q + 1] = fi;
          }
      }
      if (bot_better(v, s, bv[BOTK - 1], bi[BOTK - 1])) {
        bv[BOTK - 1] = v; bi[BOTK - 1] = s;
#pragma unroll
        for (int q = BOTK - 2; q >= 0; --q)
          if (bot_better(bv[q + 1], bi[q + 1], bv[q], bi[q])) {
            float fv = bv[q]; int fi = bi[q];
            bv[q] = bv[q + 1]; bi[q] = bi[q + 1];
            bv[q + 1] = fv;    bi[q + 1] = fi;
          }
      }
    }
    asm volatile("s_wait_dscnt 0" ::: "memory");  // loads done before next-iter stores
  }

  // Publish partial lists.
  const int p = w * 2 + hi;
#pragma unroll
  for (int q = 0; q < TOPK; ++q) { sTopV[m][p * TOPK + q] = tv[q]; sTopI[m][p * TOPK + q] = ti[q]; }
#pragma unroll
  for (int q = 0; q < BOTK; ++q) { sBotV[m][p * BOTK + q] = bv[q]; sBotI[m][p * BOTK + q] = bi[q]; }
  __syncthreads();

  // Exact merge: one thread per row.
  if (tid < 16) {
    const int r = tid;
    int ns = 0;
    for (int k = 0; k < 8; ++k) {
      float bvv = -1e30f; int bii = 0x7fffffff; int slot = -1;
      for (int q = 0; q < NPART * TOPK; ++q)
        if (top_better(sTopV[r][q], sTopI[r][q], bvv, bii)) {
          bvv = sTopV[r][q]; bii = sTopI[r][q]; slot = q;
        }
      if (bvv <= -1e29f) break;
      sSel[r][ns++] = bii;
      sTopV[r][slot] = -1e30f; sTopI[r][slot] = 0x7fffffff;
    }
    sNS[r] = ns;
    for (int q = 0; q < NPART * BOTK; ++q) {      // exclude chosen top indices
      const int i2 = sBotI[r][q];
      for (int k = 0; k < ns; ++k)
        if (i2 == sSel[r][k]) { sBotV[r][q] = 1e30f; sBotI[r][q] = 0x7fffffff; }
    }
    int nc = 0;
    for (int k = 0; k < 4; ++k) {
      float bvv = 1e30f; int bii = 0x7fffffff; int slot = -1;
      for (int q = 0; q < NPART * BOTK; ++q)
        if (bot_better(sBotV[r][q], sBotI[r][q], bvv, bii)) {
          bvv = sBotV[r][q]; bii = sBotI[r][q]; slot = q;
        }
      if (bvv >= 1e29f) break;
      sSel[r][8 + nc++] = bii;
      sBotV[r][slot] = 1e30f; sBotI[r][slot] = 0x7fffffff;
    }
    sNC[r] = nc;
  }
  __syncthreads();

  // Aggregate + pointwise: thread -> (row r, feature chunk c*16..c*16+15).
  const int r = tid & 15;
  const int c = tid >> 4;
  const int t = t0 + r;
  const int ns = sNS[r], nc = sNC[r];
  float pos[16], neg[16];
#pragma unroll
  for (int i = 0; i < 16; ++i) { pos[i] = 0.f; neg[i] = 0.f; }
  const float* xb = x + (size_t)b * TT * DD;
  for (int k = 0; k < ns; ++k) {
    const float4* xr = (const float4*)(xb + (size_t)sSel[r][k] * DD + c * 16);
#pragma unroll
    for (int q = 0; q < 4; ++q) {
      float4 v4 = xr[q];
      pos[4*q] += v4.x; pos[4*q+1] += v4.y; pos[4*q+2] += v4.z; pos[4*q+3] += v4.w;
    }
  }
  for (int k = 0; k < nc; ++k) {
    const float4* xr = (const float4*)(xb + (size_t)sSel[r][8 + k] * DD + c * 16);
#pragma unroll
    for (int q = 0; q < 4; ++q) {
      float4 v4 = xr[q];
      neg[4*q] += v4.x; neg[4*q+1] += v4.y; neg[4*q+2] += v4.z; neg[4*q+3] += v4.w;
    }
  }
  const float mix   = 1.f / (1.f + expf(-log_mix[0]));
  const float alpha = 1.f / (1.f + expf(-log_alpha[0]));
  const float scale = log1pf(expf(log_scale[0])) + 0.01f;
  const float ip = 1.f / (float)(ns > 0 ? ns : 1);
  const float ic = 1.f / (float)(nc > 0 ? nc : 1);
  const float* xt = xb + (size_t)t * DD + c * 16;
  float* op = out + ((size_t)(b * TT + t)) * DD + c * 16;
#pragma unroll
  for (int i = 0; i < 16; ++i) {
    const int d = c * 16 + i;
    const float ctx = alpha * pos[i] * ip + (1.f - alpha) * neg[i] * ic;
    const float bl  = mix * xt[i] + (1.f - mix) * ctx;
    const float y   = bl * gain[d] + bias[d];
    const float g   = 0.5f * y * (1.f + erff(y * 0.70710678118654752f));
    op[i] = g * scale;
  }
}

extern "C" void kernel_launch(void* const* d_in, const int* in_sizes, int n_in,
                              void* d_out, int out_size, void* d_ws, size_t ws_size,
                              hipStream_t stream) {
  const float* x      = (const float*)d_in[0];
  const float* gain   = (const float*)d_in[1];
  const float* bias   = (const float*)d_in[2];
  const float* lmix   = (const float*)d_in[3];
  const float* lalpha = (const float*)d_in[4];
  const float* lscale = (const float*)d_in[5];
  float* out = (float*)d_out;
  _Float16* xn = (_Float16*)d_ws;            // B*T*D*2 = 4 MiB scratch

  dgn_normalize<<<(BB * TT) / 8, 256, 0, stream>>>(x, xn);
  dgn_main<<<BB * (TT / 16), 256, 0, stream>>>(x, xn, gain, bias,
                                               lmix, lalpha, lscale, out);
}
